// MultiHeadSelfAttention_81827716923482
// MI455X (gfx1250) — matrix operand order repaired and checked by reading, not compiled, not run on hardware
//
#include <hip/hip_runtime.h>
#include <hip/hip_bf16.h>

namespace {

constexpr int kT = 2048;
constexpr int kC = 2048;
constexpr int kH = 16;
constexpr int kD = 128;
constexpr int kB = 4;
constexpr int k3C = 3 * kC;
constexpr int kM = kB * kT;   // 8192 GEMM rows

typedef __attribute__((ext_vector_type(16))) __bf16 bf16x16;
typedef __attribute__((ext_vector_type(8)))  __bf16 bf16x8;
typedef __attribute__((ext_vector_type(8)))  float  f32x8;

__device__ __forceinline__ f32x8 wmma_bf16(bf16x16 a, bf16x16 b, f32x8 c) {
  // D = A(16x32 bf16) * B(32x16 bf16) + C(16x16 f32)
  return __builtin_amdgcn_wmma_f32_16x16x32_bf16(false, a, false, b, (short)0, c,
                                                 false, false);
}

// gfx1250 async global->LDS copy (ASYNCcnt-tracked, no VGPR staging).
__device__ __forceinline__ void async_copy_b128(unsigned lds_off, const void* g) {
  asm volatile("global_load_async_to_lds_b128 %0, %1, off"
               :: "v"(lds_off), "v"(g) : "memory");
}
__device__ __forceinline__ void wait_async0() {
  asm volatile("s_wait_asynccnt 0" ::: "memory");
}
__device__ __forceinline__ unsigned lds_off(const void* p) {
  // Generic pointers into the LDS aperture carry the LDS byte address in [31:0].
  return (unsigned)(uintptr_t)p;
}

// A-fragment (16x32 bf16) from row-major [M][K] storage with element stride.
// Lanes 0-15: row=lane, K={0..7,16..23}; lanes 16-31: row=lane-16, K={8..15,24..31}.
__device__ __forceinline__ bf16x16 frag_a(const __bf16* base, int stride) {
  const int lane = threadIdx.x & 31;
  const int half = lane >> 4;
  const int r = lane & 15;
  const __bf16* p = base + r * stride + half * 8;
  union { bf16x16 v; bf16x8 h[2]; } u;
  u.h[0] = *reinterpret_cast<const bf16x8*>(p);
  u.h[1] = *reinterpret_cast<const bf16x8*>(p + 16);
  return u.v;
}

// B-fragment (32x16 bf16) from K-transposed [N][K] storage with element stride.
// Lanes 0-15: N=lane, K={0..7,16..23}; lanes 16-31: N=lane-16, K={8..15,24..31}.
__device__ __forceinline__ bf16x16 frag_b(const __bf16* base, int stride) {
  const int lane = threadIdx.x & 31;
  const int half = lane >> 4;
  const int r = lane & 15;
  const __bf16* p = base + r * stride + half * 8;
  union { bf16x16 v; bf16x8 h[2]; } u;
  u.h[0] = *reinterpret_cast<const bf16x8*>(p);
  u.h[1] = *reinterpret_cast<const bf16x8*>(p + 16);
  return u.v;
}

__global__ __launch_bounds__(256)
void mhsa_cvt_f32_bf16(const float* __restrict__ in, __bf16* __restrict__ out, int n) {
  int i = blockIdx.x * blockDim.x + threadIdx.x;
  int stride = gridDim.x * blockDim.x;
  for (; i < n; i += stride) out[i] = (__bf16)in[i];
}

// One-time weight prep: fp32 [K][N] -> bf16 [N][K] via LDS 64x64 tile transpose.
__global__ __launch_bounds__(256)
void mhsa_cvt_transpose(const float* __restrict__ in, __bf16* __restrict__ out,
                        int K, int N) {
  __shared__ float tile[64][65];
  const int k0 = blockIdx.y * 64;
  const int n0 = blockIdx.x * 64;
#pragma unroll
  for (int it = 0; it < 16; ++it) {
    int idx = threadIdx.x + it * 256;
    int r = idx >> 6, c = idx & 63;
    tile[r][c] = in[(size_t)(k0 + r) * N + n0 + c];
  }
  __syncthreads();
#pragma unroll
  for (int it = 0; it < 16; ++it) {
    int idx = threadIdx.x + it * 256;
    int r = idx >> 6, c = idx & 63;   // r: output N-row, c: K column
    out[(size_t)(n0 + r) * K + k0 + c] = (__bf16)tile[c][r];
  }
}

// ------------- Tiled bf16 GEMM: C[M][N] = A[M][K] * Bt[N][K]^T + bias -------------
// Block tile 128x256: 8 waves (2 along M x 4 along N), each wave a 64x64 tile
// (4x4 fragments, 16 WMMAs per 32-deep k-step; 8 fragment loads -> 16B/lane/WMMA).
// Double-buffered async LDS staging. mode 0: scatter bf16 Q[BH][T][D], K[BH][T][D],
// V[BH][D][T]; mode 1: fp32 row-major output.
constexpr int TM = 128, TN = 256, TK = 32;
constexpr int LDT_S = TK + 8;   // 40 elems (80B rows: 16B aligned, conflict-spread)

__global__ __launch_bounds__(256)
void mhsa_gemm_bf16(const __bf16* __restrict__ A, const __bf16* __restrict__ Bt,
                    const float* __restrict__ bias, int N, int K, int mode,
                    __bf16* __restrict__ qo, __bf16* __restrict__ ko,
                    __bf16* __restrict__ vo, float* __restrict__ fo) {
  __shared__ __bf16 As[2][TM * LDT_S];   // 2 x 10240 B
  __shared__ __bf16 Bs[2][TN * LDT_S];   // 2 x 20480 B

  const int tid = threadIdx.x;
  const int lane = tid & 31;
  const int wave = tid >> 5;
  const int wm = (wave & 1) * 64;   // 2 waves along M
  const int wn = (wave >> 1) * 64;  // 4 waves along N
  const int bm = blockIdx.y * TM;
  const int bn = blockIdx.x * TN;

  // Async-stage one 32-deep K-slice of A (128 rows) and Bt (256 rows).
  auto stage = [&](int buf, int k0) {
#pragma unroll
    for (int it = 0; it < 2; ++it) {
      int chunk = tid + it * 256;        // 512 chunks of 8 bf16
      int row = chunk >> 2;
      int co = (chunk & 3) * 8;
      async_copy_b128(lds_off(&As[buf][row * LDT_S + co]),
                      A + (size_t)(bm + row) * K + k0 + co);
    }
#pragma unroll
    for (int it = 0; it < 4; ++it) {
      int chunk = tid + it * 256;        // 1024 chunks of 8 bf16
      int row = chunk >> 2;
      int co = (chunk & 3) * 8;
      async_copy_b128(lds_off(&Bs[buf][row * LDT_S + co]),
                      Bt + (size_t)(bn + row) * K + k0 + co);
    }
  };

  const f32x8 zf = {};
  f32x8 acc[4][4];
  for (int i = 0; i < 4; ++i)
    for (int j = 0; j < 4; ++j) acc[i][j] = zf;

  stage(0, 0);
  wait_async0();
  __syncthreads();

  int buf = 0;
  for (int k0 = 0; k0 < K; k0 += TK) {
    if (k0 + TK < K) stage(buf ^ 1, k0 + TK);   // overlap with compute below

    bf16x16 a[4];
#pragma unroll
    for (int mf = 0; mf < 4; ++mf)
      a[mf] = frag_a(&As[buf][(wm + mf * 16) * LDT_S], LDT_S);
#pragma unroll
    for (int nf = 0; nf < 4; ++nf) {
      bf16x16 b = frag_b(&Bs[buf][(wn + nf * 16) * LDT_S], LDT_S);
#pragma unroll
      for (int mf = 0; mf < 4; ++mf)
        acc[mf][nf] = wmma_bf16(a[mf], b, acc[mf][nf]);
    }

    wait_async0();
    __syncthreads();
    buf ^= 1;
  }

  // Epilogue: C VGPR r holds (row r | r+8, col=lane&15) per 16x16 fragment.
  const int hi = lane >> 4;
  const int ln = lane & 15;
#pragma unroll
  for (int mf = 0; mf < 4; ++mf) {
#pragma unroll
    for (int nf = 0; nf < 4; ++nf) {
#pragma unroll
      for (int r = 0; r < 8; ++r) {
        int row = bm + wm + mf * 16 + r + hi * 8;
        int col = bn + wn + nf * 16 + ln;
        float v = acc[mf][nf][r] + bias[col];
        if (mode == 0) {
          int b = row >> 11;            // / T
          int t = row & (kT - 1);
          int which = col >> 11;        // 0:q 1:k 2:v
          int c = col & (kC - 1);
          int h = c >> 7;               // / D
          int d = c & (kD - 1);
          __bf16 bv = (__bf16)v;
          if (which == 0) {
            qo[((size_t)(b * kH + h) * kT + t) * kD + d] = bv;
          } else if (which == 1) {
            ko[((size_t)(b * kH + h) * kT + t) * kD + d] = bv;
          } else {
            // V stored transposed per head: [BH][D][T] for async-friendly attention.
            vo[((size_t)(b * kH + h) * kD + d) * kT + t] = bv;
          }
        } else {
          fo[(size_t)row * N + col] = v;
        }
      }
    }
  }
}

// ---------------- Flash attention: per (b*h, 128-row q tile) ----------------
constexpr int QT = 128;          // q rows per block (8 waves x 16)
constexpr int KTL = 64;          // keys per tile
constexpr int VLD = KTL + 8;     // 72 elems (144B rows)
constexpr int PLD = KTL + 8;

__global__ __launch_bounds__(256)
void mhsa_attn_flash(const __bf16* __restrict__ Q, const __bf16* __restrict__ Kmat,
                     const __bf16* __restrict__ Vt, __bf16* __restrict__ O) {
  __shared__ __bf16 VT[kD * VLD];        // V tile [d][key] (global Vt already d-major)
  __shared__ __bf16 Ps[8 * 16 * PLD];    // per-wave P staging [16][64]

  const int tid = threadIdx.x;
  const int lane = tid & 31;
  const int wave = tid >> 5;
  const int hi = lane >> 4;
  const int ln = lane & 15;
  const int bh = blockIdx.x;             // 0..63
  const int qt = blockIdx.y;             // 0..15
  const int q0 = qt * QT + wave * 16;
  const size_t base = (size_t)bh * kT * kD;   // Q/K base; Vt uses [BH][D][T]

  // Q fragments for this wave's 16 rows (D=128 -> 4 A-frags), kept in VGPRs.
  bf16x16 aq[4];
#pragma unroll
  for (int f = 0; f < 4; ++f)
    aq[f] = frag_a(Q + base + (size_t)q0 * kD + f * 32, kD);

  const f32x8 zf = {};
  f32x8 oacc[8];
#pragma unroll
  for (int c = 0; c < 8; ++c) oacc[c] = zf;
  float mrow[8], lrow[8];
#pragma unroll
  for (int r = 0; r < 8; ++r) { mrow[r] = -3e38f; lrow[r] = 0.f; }

  const float scale = 0.088388347648318447f;  // 1/sqrt(128)
  const int kEnd = qt * QT + QT;

  for (int kt = 0; kt < kEnd; kt += KTL) {
    // Issue async V-tile copy (128 d-rows x 64 keys, contiguous b128 chunks);
    // overlaps with the QK^T WMMAs + softmax below, which don't touch VT.
#pragma unroll
    for (int it = 0; it < 4; ++it) {
      int chunk = tid + it * 256;          // 1024 chunks of 8 bf16
      int dr = chunk >> 3;                 // d row 0..127
      int co = (chunk & 7) * 8;            // key offset
      async_copy_b128(lds_off(&VT[dr * VLD + co]),
                      Vt + base + (size_t)dr * kT + kt + co);
    }

    // S = Q K^T (16 x 64): K row-major [key][d] == B-fragment order directly.
    f32x8 s[4];
#pragma unroll
    for (int nf = 0; nf < 4; ++nf) s[nf] = zf;
#pragma unroll
    for (int dsi = 0; dsi < 4; ++dsi) {
#pragma unroll
      for (int nf = 0; nf < 4; ++nf) {
        bf16x16 bk = frag_b(Kmat + base + (size_t)(kt + nf * 16) * kD + dsi * 32, kD);
        s[nf] = wmma_bf16(aq[dsi], bk, s[nf]);
      }
    }

    // Scale + causal mask (only on diagonal-overlapping tiles).
    const bool diag = (kt + KTL > q0);
#pragma unroll
    for (int nf = 0; nf < 4; ++nf) {
#pragma unroll
      for (int r = 0; r < 8; ++r) {
        float x = s[nf][r] * scale;
        if (diag) {
          int row = q0 + r + hi * 8;
          int col = kt + nf * 16 + ln;
          if (col > row) x = -3e38f;
        }
        s[nf][r] = x;
      }
    }

    // Online softmax; row (r | r+8) lives in a 16-lane half -> xor masks 1,2,4,8.
#pragma unroll
    for (int r = 0; r < 8; ++r) {
      float tmax = fmaxf(fmaxf(s[0][r], s[1][r]), fmaxf(s[2][r], s[3][r]));
#pragma unroll
      for (int msk = 1; msk <= 8; msk <<= 1)
        tmax = fmaxf(tmax, __shfl_xor(tmax, msk, 32));
      float nm = fmaxf(mrow[r], tmax);
      float alpha = __expf(mrow[r] - nm);
      float rsum = 0.f;
#pragma unroll
      for (int nf = 0; nf < 4; ++nf) {
        float p = __expf(s[nf][r] - nm);
        s[nf][r] = p;
        rsum += p;
      }
#pragma unroll
      for (int msk = 1; msk <= 8; msk <<= 1)
        rsum += __shfl_xor(rsum, msk, 32);
      lrow[r] = lrow[r] * alpha + rsum;
      mrow[r] = nm;
#pragma unroll
      for (int c = 0; c < 8; ++c) oacc[c][r] *= alpha;
    }

    // P (C-layout) -> per-wave LDS -> A-fragments for P@V.
    __bf16* pb = &Ps[wave * 16 * PLD];
#pragma unroll
    for (int nf = 0; nf < 4; ++nf) {
#pragma unroll
      for (int r = 0; r < 8; ++r)
        pb[(r + hi * 8) * PLD + nf * 16 + ln] = (__bf16)s[nf][r];
    }
    bf16x16 ap0 = frag_a(pb, PLD);        // keys 0..31
    bf16x16 ap1 = frag_a(pb + 32, PLD);   // keys 32..63

    wait_async0();        // V tile resident
    __syncthreads();

    // O += P * V : D=128 -> 8 N-frags, K=64 -> 2 steps.
#pragma unroll
    for (int c = 0; c < 8; ++c) {
      bf16x16 bv0 = frag_b(&VT[(c * 16) * VLD + 0], VLD);
      bf16x16 bv1 = frag_b(&VT[(c * 16) * VLD + 32], VLD);
      oacc[c] = wmma_bf16(ap0, bv0, oacc[c]);
      oacc[c] = wmma_bf16(ap1, bv1, oacc[c]);
    }
    __syncthreads();      // everyone done with VT before next tile's async writes
  }

  // Normalize and write O as [B][T][H*D] bf16 (row-major input to proj GEMM).
  const int b = bh >> 4;
  const int h = bh & 15;
#pragma unroll
  for (int r = 0; r < 8; ++r) {
    float inv = 1.0f / lrow[r];
    int row = q0 + r + hi * 8;
#pragma unroll
    for (int c = 0; c < 8; ++c)
      O[((size_t)(b * kT + row)) * kC + h * kD + c * 16 + ln] =
          (__bf16)(oacc[c][r] * inv);
  }
}

}  // namespace

extern "C" void kernel_launch(void* const* d_in, const int* in_sizes, int n_in,
                              void* d_out, int out_size, void* d_ws, size_t ws_size,
                              hipStream_t stream) {
  const float* x     = (const float*)d_in[0];
  const float* Wqkv  = (const float*)d_in[1];
  const float* bqkv  = (const float*)d_in[2];
  const float* Wproj = (const float*)d_in[3];
  const float* bproj = (const float*)d_in[4];
  float* out = (float*)d_out;

  // Workspace (bf16): x | Wqkv^T | Wproj^T | Q | K | V^T | O  (~192 MB total)
  char* ws = (char*)d_ws;
  const size_t szX  = (size_t)kM * kC * sizeof(__bf16);
  const size_t szWq = (size_t)kC * k3C * sizeof(__bf16);
  const size_t szWp = (size_t)kC * kC * sizeof(__bf16);
  const size_t nHead = (size_t)kB * kH * kT * kD;
  __bf16* xb  = (__bf16*)(ws);
  __bf16* wqt = (__bf16*)(ws + szX);                 // Wqkv^T [3C][C]
  __bf16* wpt = (__bf16*)(ws + szX + szWq);          // Wproj^T [C][C]
  __bf16* Qb  = (__bf16*)(ws + szX + szWq + szWp);
  __bf16* Kb  = Qb + nHead;
  __bf16* Vbt = Kb + nHead;                          // V^T per head [BH][D][T]
  __bf16* Ob  = Vbt + nHead;

  // 1) fp32 -> bf16; weights also transposed once so all GEMM staging is
  //    contiguous and can use global_load_async_to_lds_b128.
  mhsa_cvt_f32_bf16<<<4096, 256, 0, stream>>>(x, xb, kM * kC);
  mhsa_cvt_transpose<<<dim3(k3C / 64, kC / 64), 256, 0, stream>>>(Wqkv, wqt, kC, k3C);
  mhsa_cvt_transpose<<<dim3(kC / 64, kC / 64), 256, 0, stream>>>(Wproj, wpt, kC, kC);

  // 2) QKV projection GEMM: [8192,2048] x [2048,6144], scatter to Q/K/V^T.
  dim3 g1(k3C / TN, kM / TM);
  mhsa_gemm_bf16<<<g1, 256, 0, stream>>>(xb, wqt, bqkv, k3C, kC, 0,
                                         Qb, Kb, Vbt, nullptr);

  // 3) Causal flash attention over 64 (b,h) pairs x 16 q-tiles.
  dim3 g2(kB * kH, kT / QT);
  mhsa_attn_flash<<<g2, 256, 0, stream>>>(Qb, Kb, Vbt, Ob);

  // 4) Output projection GEMM: [8192,2048] x [2048,2048] + bias -> fp32 out.
  dim3 g3(kC / TN, kM / TM);
  mhsa_gemm_bf16<<<g3, 256, 0, stream>>>(Ob, wpt, bproj, kC, kC, 1,
                                         nullptr, nullptr, nullptr, out);
}
